// DeepGraphEmulator_53137335386501
// MI455X (gfx1250) — matrix operand order repaired and checked by reading, not compiled, not run on hardware
//
#include <hip/hip_runtime.h>

#define NNODES 20000
#define NEDGES 160000
#define NGRAPHS 4
#define NREAL 16000
#define NF 128

typedef __attribute__((ext_vector_type(16))) __bf16 v16bf;
typedef __attribute__((ext_vector_type(8)))  float  v8f;
typedef __attribute__((ext_vector_type(8)))  unsigned short ushort8;

union FragU { v16bf v; ushort8 h[2]; };
union H8   { ushort8 v; unsigned short s[8]; };

__device__ __forceinline__ unsigned short f2bf(float f) {
  union { float f; unsigned u; } x; x.f = f;
  unsigned u = x.u;
  u += 0x7fffu + ((u >> 16) & 1u);   // round-to-nearest-even
  return (unsigned short)(u >> 16);
}

__device__ __forceinline__ float activate(float v, int act, float al) {
  if (act == 1) return v > 0.f ? v : 0.f;
  if (act == 2) return v > 0.f ? v : al * v;
  return v;
}

__device__ __forceinline__ void atomAddF(float* p, float v) {
  __hip_atomic_fetch_add(p, v, __ATOMIC_RELAXED, __HIP_MEMORY_SCOPE_AGENT);
}

// convert 16 fp32 -> bf16 and store as two 16B LDS writes
__device__ __forceinline__ void cvt_store16(unsigned short* dst,
                                            float4 a, float4 b,
                                            float4 c, float4 d) {
  H8 lo, hi;
  lo.s[0]=f2bf(a.x); lo.s[1]=f2bf(a.y); lo.s[2]=f2bf(a.z); lo.s[3]=f2bf(a.w);
  lo.s[4]=f2bf(b.x); lo.s[5]=f2bf(b.y); lo.s[6]=f2bf(b.z); lo.s[7]=f2bf(b.w);
  hi.s[0]=f2bf(c.x); hi.s[1]=f2bf(c.y); hi.s[2]=f2bf(c.z); hi.s[3]=f2bf(c.w);
  hi.s[4]=f2bf(d.x); hi.s[5]=f2bf(d.y); hi.s[6]=f2bf(d.z); hi.s[7]=f2bf(d.w);
  *reinterpret_cast<ushort8*>(dst)     = lo.v;
  *reinterpret_cast<ushort8*>(dst + 8) = hi.v;
}

// ---------------------------------------------------------------------------
// Fused up-to-3-layer MLP: out = L3(act2(L2(act1(L1(concat(srcs))))))
// All hidden/output widths are 128. Layer-1 input is a (gather-)concat of up
// to 3 fp32 sources. Intermediates stay in LDS as bf16.
// Tile: 128x128 per workgroup; 8 waves in a 4(M)x2(N) grid, each wave a
// 32x64 block = 2x4 WMMA acc tiles (12 ds_load_b128 per 8 WMMAs).
// ---------------------------------------------------------------------------
__global__ __launch_bounds__(256)
void fused_mlp_kernel(
    const float* __restrict__ s0, int k0n, const int* __restrict__ i0,
    const float* __restrict__ s1, int k1n, const int* __restrict__ i1,
    const float* __restrict__ s2, int k2n, const int* __restrict__ i2,
    const float* __restrict__ W0, const float* __restrict__ B0, int a0, const float* __restrict__ al0,
    const float* __restrict__ W1, const float* __restrict__ B1, int a1, const float* __restrict__ al1,
    const float* __restrict__ W2, const float* __restrict__ B2, int a2, const float* __restrict__ al2,
    int nlayers, int M, float* __restrict__ out)
{
  __shared__ __align__(16) unsigned short As[128 * 40];   // A tile (32 K + pad)
  __shared__ __align__(16) unsigned short Bs[128 * 40];   // B tile (W rows)
  __shared__ __align__(16) unsigned short Hs[128 * 136];  // inter-layer acts

  const int tid  = threadIdx.x;
  const int wave = tid >> 5;
  const int lane = tid & 31;
  const int m0   = blockIdx.x * 128;

  const int Ktot = k0n + k1n + k2n;

  // staging mapping: 2 threads per row, 16 contiguous K each
  const int lrow = tid >> 1;
  const int lj0  = (tid & 1) << 4;

  // fragment mapping (ISA layouts); wave tile = rows [32*wm,+32) x cols [64*wn,+64)
  const int sel  = lane >> 4;                 // 0 or 1
  const int wm   = wave & 3;                  // M-block 0..3
  const int wn   = wave >> 2;                 // N-block 0..1
  const int ka   = sel << 3;                  // A: K 0..7 / 8..15 (+16 for hi)
  const int kb   = sel << 3;                  // B: K 0..7 / 8..15 (+16 for hi)
  const int ccol = lane & 15;
  const int lrow16 = lane & 15;

  // per-thread gather row bases for layer-0 A staging (hoisted out of K loop)
  const int gm = m0 + lrow;
  const float* ab0 = nullptr; const float* ab1 = nullptr; const float* ab2 = nullptr;
  if (gm < M) {
    if (s0) ab0 = s0 + (size_t)(i0 ? i0[gm] : gm) * (size_t)k0n;
    if (s1) ab1 = s1 + (size_t)(i1 ? i1[gm] : gm) * (size_t)k1n;
    if (s2) ab2 = s2 + (size_t)(i2 ? i2[gm] : gm) * (size_t)k2n;
  }

  for (int layer = 0; layer < nlayers; ++layer) {
    const float* W   = (layer == 0) ? W0 : (layer == 1 ? W1 : W2);
    const float* Bb  = (layer == 0) ? B0 : (layer == 1 ? B1 : B2);
    const int    act = (layer == 0) ? a0 : (layer == 1 ? a1 : a2);
    const float* alp = (layer == 0) ? al0 : (layer == 1 ? al1 : al2);
    const int K      = (layer == 0) ? Ktot : NF;
    const int ksteps = (K + 31) >> 5;
    const float* wrow = W + (size_t)lrow * (size_t)K;

    v8f acc[8];
#pragma unroll
    for (int t = 0; t < 8; ++t) acc[t] = {};

    for (int ks = 0; ks < ksteps; ++ks) {
      const int kbase = ks << 5;
      const int krow  = kbase + lj0;
      // ---- stage B (weight) tile: Bs[n][kk] = W[n][kbase+kk] ----
      {
        unsigned short* dst = &Bs[lrow * 40 + lj0];
        if (krow + 16 <= K) {               // fast path: K-aligned block
          const float4* p4 = reinterpret_cast<const float4*>(wrow + krow);
          cvt_store16(dst, p4[0], p4[1], p4[2], p4[3]);
        } else {                            // ragged (tiny-K encoder layer)
          H8 lo, hi;
#pragma unroll
          for (int jj = 0; jj < 8; ++jj)
            lo.s[jj] = f2bf((krow + jj < K) ? wrow[krow + jj] : 0.f);
#pragma unroll
          for (int jj = 0; jj < 8; ++jj)
            hi.s[jj] = f2bf((krow + 8 + jj < K) ? wrow[krow + 8 + jj] : 0.f);
          *reinterpret_cast<ushort8*>(dst)     = lo.v;
          *reinterpret_cast<ushort8*>(dst + 8) = hi.v;
        }
      }
      // ---- stage A tile (layer 0 only): gather + concat + cvt ----
      if (layer == 0) {
        const int k = krow;
        const float* ab; int kk, Ks;
        if (k < k0n)            { ab = ab0; kk = k;             Ks = k0n; }
        else if (k < k0n + k1n) { ab = ab1; kk = k - k0n;       Ks = k1n; }
        else                    { ab = ab2; kk = k - k0n - k1n; Ks = k2n; }
        unsigned short* dst = &As[lrow * 40 + lj0];
        if (ab && kk + 16 <= Ks) {          // fast path
          const float4* p4 = reinterpret_cast<const float4*>(ab + kk);
          cvt_store16(dst, p4[0], p4[1], p4[2], p4[3]);
        } else {                            // ragged / out-of-range -> zeros
          H8 lo, hi;
#pragma unroll
          for (int jj = 0; jj < 8; ++jj)
            lo.s[jj] = f2bf((ab && kk + jj < Ks) ? ab[kk + jj] : 0.f);
#pragma unroll
          for (int jj = 0; jj < 8; ++jj)
            hi.s[jj] = f2bf((ab && kk + 8 + jj < Ks) ? ab[kk + 8 + jj] : 0.f);
          *reinterpret_cast<ushort8*>(dst)     = lo.v;
          *reinterpret_cast<ushort8*>(dst + 8) = hi.v;
        }
      }
      __syncthreads();
      // ---- fragments + WMMA (2 A-frags x 4 B-frags) ----
      FragU a[2];
#pragma unroll
      for (int i = 0; i < 2; ++i) {
        const int frow = (wm << 5) + (i << 4) + lrow16;
        if (layer == 0) {
          a[i].h[0] = *reinterpret_cast<const ushort8*>(&As[frow * 40 + ka]);
          a[i].h[1] = *reinterpret_cast<const ushort8*>(&As[frow * 40 + 16 + ka]);
        } else {
          a[i].h[0] = *reinterpret_cast<const ushort8*>(&Hs[frow * 136 + kbase + ka]);
          a[i].h[1] = *reinterpret_cast<const ushort8*>(&Hs[frow * 136 + kbase + 16 + ka]);
        }
      }
      FragU bf[4];
#pragma unroll
      for (int j = 0; j < 4; ++j) {
        const int col = (wn << 6) + (j << 4) + ccol;
        bf[j].h[0] = *reinterpret_cast<const ushort8*>(&Bs[col * 40 + kb]);
        bf[j].h[1] = *reinterpret_cast<const ushort8*>(&Bs[col * 40 + kb + 16]);
      }
#pragma unroll
      for (int i = 0; i < 2; ++i)
#pragma unroll
        for (int j = 0; j < 4; ++j)
          acc[i * 4 + j] = __builtin_amdgcn_wmma_f32_16x16x32_bf16(
              false, a[i].v, false, bf[j].v, (short)0, acc[i * 4 + j], false, false);
      __syncthreads();
    }

    // ---- epilogue ----
    float alpha = 0.f;
    if (act == 2 && alp) alpha = *alp;
    if (layer == nlayers - 1) {
#pragma unroll
      for (int i = 0; i < 2; ++i) {
        const int rb = (wm << 5) + (i << 4) + (sel << 3);
#pragma unroll
        for (int j = 0; j < 4; ++j) {
          const int gcol = (wn << 6) + (j << 4) + ccol;
          const float bv = Bb ? Bb[gcol] : 0.f;
#pragma unroll
          for (int r = 0; r < 8; ++r) {
            const int grow = m0 + rb + r;
            if (grow < M) {
              float v = activate(acc[i * 4 + j][r] + bv, act, alpha);
              out[(size_t)grow * NF + gcol] = v;
            }
          }
        }
      }
    } else {
#pragma unroll
      for (int i = 0; i < 2; ++i) {
        const int rb = (wm << 5) + (i << 4) + (sel << 3);
#pragma unroll
        for (int j = 0; j < 4; ++j) {
          const int gcol = (wn << 6) + (j << 4) + ccol;
          const float bv = Bb ? Bb[gcol] : 0.f;
#pragma unroll
          for (int r = 0; r < 8; ++r) {
            float v = activate(acc[i * 4 + j][r] + bv, act, alpha);
            Hs[(rb + r) * 136 + gcol] = f2bf(v);
          }
        }
      }
      __syncthreads();
    }
  }
}

// ---------------------------------------------------------------------------
// Small helper kernels
// ---------------------------------------------------------------------------
__global__ void add_v_kernel(float* __restrict__ h, const float* __restrict__ v,
                             const int* __restrict__ batch, long long total) {
  long long i = (long long)blockIdx.x * blockDim.x + threadIdx.x;
  if (i >= total) return;
  int n = (int)(i >> 7), c = (int)(i & 127);
  h[i] += v[batch[n] * NF + c];
}

__global__ void ew_add_kernel(float* __restrict__ a, const float* __restrict__ b,
                              long long total) {
  long long i = (long long)blockIdx.x * blockDim.x + threadIdx.x;
  if (i < total) a[i] += b[i];
}

__global__ void scatter_edge_kernel(const float* __restrict__ msg,
                                    const int* __restrict__ snd,
                                    const int* __restrict__ rcv,
                                    float* __restrict__ outb, int E) {
  long long i = (long long)blockIdx.x * blockDim.x + threadIdx.x;
  if (i >= (long long)E * NF) return;
  int e = (int)(i >> 7), c = (int)(i & 127);
  float m = msg[i];
  atomAddF(&outb[(size_t)rcv[e] * NF + c],  m);
  atomAddF(&outb[(size_t)snd[e] * NF + c], -m);
}

__global__ __launch_bounds__(128)
void pool_kernel(const float* __restrict__ h, const int* __restrict__ batch,
                 float* __restrict__ pooled, int N) {
  int col = threadIdx.x;           // 0..127
  int n0  = blockIdx.x * 256;
  float a0 = 0.f, a1 = 0.f, a2 = 0.f, a3 = 0.f;
  for (int j = 0; j < 256; ++j) {
    int n = n0 + j;
    if (n >= N) break;
    int g = batch[n];
    float v = h[(size_t)n * NF + col];
    a0 += (g == 0) ? v : 0.f;
    a1 += (g == 1) ? v : 0.f;
    a2 += (g == 2) ? v : 0.f;
    a3 += (g == 3) ? v : 0.f;
  }
  if (a0 != 0.f) atomAddF(&pooled[0 * NF + col], a0);
  if (a1 != 0.f) atomAddF(&pooled[1 * NF + col], a1);
  if (a2 != 0.f) atomAddF(&pooled[2 * NF + col], a2);
  if (a3 != 0.f) atomAddF(&pooled[3 * NF + col], a3);
}

__global__ __launch_bounds__(256)
void vn_mlp_kernel(const float* __restrict__ pooled,
                   const float* __restrict__ W0, const float* __restrict__ b0,
                   const float* __restrict__ W1, const float* __restrict__ b1,
                   float* __restrict__ v) {
  __shared__ float t[4 * 256];
  int tid = threadIdx.x;
  for (int g = 0; g < 4; ++g) {
    float s = b0[tid];
    for (int k = 0; k < 128; ++k) s += pooled[g * 128 + k] * W0[tid * 128 + k];
    t[g * 256 + tid] = s > 0.f ? s : 0.f;
  }
  __syncthreads();
  if (tid < 128) {
    for (int g = 0; g < 4; ++g) {
      float s = b1[tid];
      for (int k = 0; k < 256; ++k) s += t[g * 256 + k] * W1[tid * 256 + k];
      v[g * 128 + tid] = s > 0.f ? s : 0.f;
    }
  }
}

__global__ void bcast_v_kernel(float* __restrict__ v, const float* __restrict__ emb) {
  int i = blockIdx.x * blockDim.x + threadIdx.x;
  if (i < NGRAPHS * NF) v[i] = emb[i & 127];
}

__global__ __launch_bounds__(256)
void dec_out_kernel(const float* __restrict__ hdec, const float* __restrict__ W,
                    const float* __restrict__ b, float* __restrict__ out,
                    int Nreal, int comp) {
  int wave = threadIdx.x >> 5, lane = threadIdx.x & 31;
  int row = blockIdx.x * 8 + wave;
  if (row >= Nreal) return;
  float s = 0.f;
  for (int k = lane; k < 128; k += 32) s += hdec[(size_t)row * NF + k] * W[k];
  for (int off = 16; off; off >>= 1) s += __shfl_xor(s, off, 32);
  if (lane == 0) out[(size_t)row * 3 + comp] = s + b[0];
}

// ---------------------------------------------------------------------------
// Host orchestration
// ---------------------------------------------------------------------------
struct EncP { const float *W0,*b0,*W1,*b1,*W2,*b2,*p0,*p1; };
struct BlkP {
  const float *e0W,*e0b,*e1W,*e1b,*e2W,*e2b;
  const float *g0W,*g0b,*g1W,*g1b,*g2W,*g2b;
  const float *n0W,*n0b,*n1W,*n1b,*n2W,*n2b;
  const float *emb,*v0W,*v0b,*v1W,*v1b;
};

extern "C" void kernel_launch(void* const* d_in, const int* in_sizes, int n_in,
                              void* d_out, int out_size, void* d_ws, size_t ws_size,
                              hipStream_t stream) {
  (void)in_sizes; (void)n_in; (void)out_size; (void)ws_size;

  const float* X   = (const float*)d_in[0];   // [20000,6]
  const float* EAi = (const float*)d_in[1];   // [160000,4]
  const float* U   = (const float*)d_in[2];   // [4,8]
  const int*   snd = (const int*)d_in[3];
  const int*   rcv = (const int*)d_in[4];
  const int*   bat = (const int*)d_in[5];
  const int*   rn  = (const int*)d_in[6];

  auto F = [&](int i) { return (const float*)d_in[i]; };
  const int PB = 7;
  auto getEnc = [&](int base) {
    EncP e;
    e.W0=F(base+0); e.b0=F(base+1); e.W1=F(base+2); e.b1=F(base+3);
    e.W2=F(base+4); e.b2=F(base+5); e.p0=F(base+6); e.p1=F(base+7);
    return e;
  };
  auto getBlk = [&](int base) {
    BlkP b; int i = base;
    b.e0W=F(i++); b.e0b=F(i++); b.e1W=F(i++); b.e1b=F(i++); b.e2W=F(i++); b.e2b=F(i++);
    b.g0W=F(i++); b.g0b=F(i++); b.g1W=F(i++); b.g1b=F(i++); b.g2W=F(i++); b.g2b=F(i++);
    b.n0W=F(i++); b.n0b=F(i++); b.n1W=F(i++); b.n1b=F(i++); b.n2W=F(i++); b.n2b=F(i++);
    b.emb=F(i++); b.v0W=F(i++); b.v0b=F(i++); b.v1W=F(i++); b.v1b=F(i++);
    return b;
  };
  // jax pytree order (dict keys sorted): dec_x, dec_y, dec_z, enc_e, enc_u,
  // enc_x, xx[0], xx[1], yy[0], yy[1], zz[0], zz[1]
  EncP decs[3] = { getEnc(PB + 0), getEnc(PB + 8), getEnc(PB + 16) };
  EncP enc_e = getEnc(PB + 24), enc_u = getEnc(PB + 32), enc_x = getEnc(PB + 40);
  int gbase[3] = { PB + 48, PB + 48 + 46, PB + 48 + 92 };

  // workspace layout (fp32)
  const size_t NN128 = (size_t)NNODES * NF;      // 2.56M
  const size_t NE128 = (size_t)NEDGES * NF;      // 20.48M
  float* ws  = (float*)d_ws;
  float* h   = ws;
  float* nb1 = h   + NN128;   // messages_u / dec temp
  float* nb2 = nb1 + NN128;   // gx
  float* nb3 = nb2 + NN128;   // received
  float* ea  = nb3 + NN128;
  float* eb1 = ea  + NE128;   // messages
  float* ue  = eb1 + NE128;   // u embedding [4,128]
  float* vvb = ue  + 512;     // v           [4,128]
  float* pl  = vvb + 512;     // pooled      [4,128]

  auto mlp = [&](const float* s0, int k0, const int* i0,
                 const float* s1, int k1, const int* i1,
                 const float* s2, int k2, const int* i2,
                 const float* w0, const float* bb0, int a0c, const float* al0,
                 const float* w1, const float* bb1, int a1c, const float* al1,
                 const float* w2, const float* bb2, int a2c, const float* al2,
                 int nl, int M, float* outp) {
    dim3 grid((M + 127) / 128);
    fused_mlp_kernel<<<grid, dim3(256), 0, stream>>>(
        s0, k0, i0, s1, k1, i1, s2, k2, i2,
        w0, bb0, a0c, al0, w1, bb1, a1c, al1, w2, bb2, a2c, al2,
        nl, M, outp);
  };

  const int TPB = 256;
  auto nblk = [&](long long n) { return (unsigned)((n + TPB - 1) / TPB); };

  // u embedding (once)
  mlp(U, 8, nullptr, nullptr, 0, nullptr, nullptr, 0, nullptr,
      enc_u.W0, enc_u.b0, 2, enc_u.p0,
      enc_u.W1, enc_u.b1, 2, enc_u.p1,
      enc_u.W2, enc_u.b2, 0, nullptr, 3, NGRAPHS, ue);

  for (int gi = 0; gi < 3; ++gi) {
    // (re)encode node + edge features into working buffers (cheap GEMMs)
    mlp(X, 6, nullptr, nullptr, 0, nullptr, nullptr, 0, nullptr,
        enc_x.W0, enc_x.b0, 2, enc_x.p0,
        enc_x.W1, enc_x.b1, 2, enc_x.p1,
        enc_x.W2, enc_x.b2, 0, nullptr, 3, NNODES, h);
    mlp(EAi, 4, nullptr, nullptr, 0, nullptr, nullptr, 0, nullptr,
        enc_e.W0, enc_e.b0, 2, enc_e.p0,
        enc_e.W1, enc_e.b1, 2, enc_e.p1,
        enc_e.W2, enc_e.b2, 0, nullptr, 3, NEDGES, ea);

    for (int k = 0; k < 2; ++k) {
      BlkP b = getBlk(gbase[gi] + k * 23);
      if (k == 0)
        bcast_v_kernel<<<nblk(NGRAPHS * NF), TPB, 0, stream>>>(vvb, b.emb);
      add_v_kernel<<<nblk(NN128), TPB, 0, stream>>>(h, vvb, bat, (long long)NN128);

      // edge MLP: concat(ea, h[snd], h[rcv]) -> messages (eb1)
      mlp(ea, 128, nullptr, h, 128, snd, h, 128, rcv,
          b.e0W, b.e0b, 1, nullptr, b.e1W, b.e1b, 1, nullptr,
          b.e2W, b.e2b, 0, nullptr, 3, NEDGES, eb1);
      // global MLP: concat(h, u[batch]) -> messages_u (nb1)
      mlp(h, 128, nullptr, ue, 128, bat, nullptr, 0, nullptr,
          b.g0W, b.g0b, 1, nullptr, b.g1W, b.g1b, 1, nullptr,
          b.g2W, b.g2b, 0, nullptr, 3, NNODES, nb1);
      // received = seg(msg, rcv) - seg(msg, snd)  (nb3)
      hipMemsetAsync(nb3, 0, NN128 * sizeof(float), stream);
      scatter_edge_kernel<<<nblk(NE128), TPB, 0, stream>>>(eb1, snd, rcv, nb3, NEDGES);
      // node MLP: concat(h, received, messages_u) -> gx (nb2)
      mlp(h, 128, nullptr, nb3, 128, nullptr, nb1, 128, nullptr,
          b.n0W, b.n0b, 1, nullptr, b.n1W, b.n1b, 1, nullptr,
          b.n2W, b.n2b, 0, nullptr, 3, NNODES, nb2);
      // residuals
      ew_add_kernel<<<nblk(NN128), TPB, 0, stream>>>(h, nb2, (long long)NN128);
      ew_add_kernel<<<nblk(NE128), TPB, 0, stream>>>(ea, eb1, (long long)NE128);
      // pooled = segsum(h, batch) + v ; v = vn_mlp(pooled)
      hipMemcpyAsync(pl, vvb, 512 * sizeof(float), hipMemcpyDeviceToDevice, stream);
      pool_kernel<<<(NNODES + 255) / 256, 128, 0, stream>>>(h, bat, pl, NNODES);
      vn_mlp_kernel<<<1, 256, 0, stream>>>(pl, b.v0W, b.v0b, b.v1W, b.v1b, vvb);
    }

    // decoder: two PReLU layers on gathered rows, then 128->1 dot
    const EncP& D = decs[gi];
    mlp(h, 128, rn, nullptr, 0, nullptr, nullptr, 0, nullptr,
        D.W0, D.b0, 2, D.p0, D.W1, D.b1, 2, D.p1,
        nullptr, nullptr, 0, nullptr, 2, NREAL, nb1);
    dec_out_kernel<<<(NREAL + 7) / 8, 256, 0, stream>>>(
        nb1, D.W2, D.b2, (float*)d_out, NREAL, gi);
  }
}
